// Decoder_80779744903970
// MI455X (gfx1250) — matrix operand order repaired and checked by reading, not compiled, not run on hardware
//
#include <hip/hip_runtime.h>
#include <stdint.h>

// ---------------------------------------------------------------------------
// LSTM decoder for MI455X (gfx1250, wave32, WMMA), persistent-kernel version.
//   gates = [h | x_t] @ [W_hh | W_ih]^T   (bf16 operands, f32 accumulate)
// One persistent kernel runs all 511 steps; a device-scope generation barrier
// (16 co-resident blocks) carries the recurrence. Each wave owns one 16-row x
// 16-col tile in all four gate quadrants (i,f,g,o), so the activation math is
// register-resident and no gates buffer exists. Working set (~50 MB) is
// L2-resident (192 MB). A-tile staging uses the CDNA5 async global->LDS path
// (ASYNCcnt) when the toolchain exposes it.
// ---------------------------------------------------------------------------

typedef __attribute__((ext_vector_type(16))) __bf16 v16bf;
typedef __attribute__((ext_vector_type(2)))  __bf16 v2bf;
typedef __attribute__((ext_vector_type(8)))  float  v8f;
typedef __attribute__((ext_vector_type(4)))  int    v4i;

static constexpr int B  = 32;
static constexpr int T  = 512;
static constexpr int TP = 511;          // T-1 steps
static constexpr int H  = 1024;
static constexpr int G  = 4096;         // 4H
static constexpr int K2 = 2048;         // fused K = 2H
static constexpr int NB = 16;           // persistent blocks

// CDNA5 async global->LDS path (gated so compile never breaks)
#if defined(__has_builtin)
#if __has_builtin(__builtin_amdgcn_global_load_async_to_lds_b128) && \
    __has_builtin(__builtin_amdgcn_s_wait_asynccnt)
#define USE_ASYNC_LDS 1
#endif
#endif
#ifndef USE_ASYNC_LDS
#define USE_ASYNC_LDS 0
#endif

// workspace layout (bytes, 256-aligned)
static constexpr size_t OFF_WCAT  = 0;                           // bf16 [4096][2048]
static constexpr size_t OFF_EMB   = OFF_WCAT + (size_t)G*K2*2;   // bf16 [511][32][1024]
static constexpr size_t OFF_BIAS  = OFF_EMB  + (size_t)TP*B*H*2; // f32 [4096]
static constexpr size_t OFF_HBF   = OFF_BIAS + (size_t)G*4;      // bf16 [32][1024]
static constexpr size_t OFF_C     = OFF_HBF  + (size_t)B*H*2;    // f32 [32][1024]
static constexpr size_t OFF_BAR   = OFF_C    + (size_t)B*H*4;    // u32 [2] barrier

union Pack2 { uint32_t u; __bf16 h[2]; };
union AFrag { v16bf v; v2bf p[8]; };

__device__ __forceinline__ float fsigmoid(float x) {
  return 1.0f / (1.0f + __expf(-x));
}
__device__ __forceinline__ float ftanh(float x) {
  float e = __expf(2.0f * x);
  return (e - 1.0f) / (e + 1.0f);
}

// --- prep: Wcat_bf16[n][k] = (k<H ? W_hh : W_ih)[n][k%H] ---------------------
__global__ void prep_wcat(const float* __restrict__ W_ih,
                          const float* __restrict__ W_hh,
                          uint32_t* __restrict__ wcat_w) {
  int p = blockIdx.x * blockDim.x + threadIdx.x;      // one bf16 pair each
  if (p >= G * (K2 / 2)) return;
  int n  = p >> 10;
  int kp = p & 1023;
  int k  = kp << 1;
  const float* src = (k < H) ? (W_hh + (size_t)n * H + k)
                             : (W_ih + (size_t)n * H + (k - H));
  Pack2 o; o.h[0] = (__bf16)src[0]; o.h[1] = (__bf16)src[1];
  wcat_w[p] = o.u;
}

// --- prep: gather embeddings to bf16, layout [t][b][H] -----------------------
__global__ void prep_emb(const int* __restrict__ tgt,       // [B][T]
                         const float* __restrict__ emb_table,
                         uint32_t* __restrict__ emb_w) {
  int p = blockIdx.x * blockDim.x + threadIdx.x;      // one bf16 pair each
  if (p >= TP * B * (H / 2)) return;
  int t   = p / (B * (H / 2));
  int rem = p - t * (B * (H / 2));
  int b   = rem >> 9;
  int kp  = rem & 511;
  int idx = tgt[(size_t)b * T + t];
  const float* src = emb_table + (size_t)idx * H + (kp << 1);
  Pack2 o; o.h[0] = (__bf16)src[0]; o.h[1] = (__bf16)src[1];
  emb_w[p] = o.u;
}

// --- prep: bias = b_ih + b_hh ; state + barrier init -------------------------
__global__ void prep_state(const float* __restrict__ h0,
                           const float* __restrict__ c0,
                           const float* __restrict__ b_ih,
                           const float* __restrict__ b_hh,
                           float* __restrict__ bias,
                           __bf16* __restrict__ h_bf,
                           float* __restrict__ c_st,
                           uint32_t* __restrict__ bar) {
  int i = blockIdx.x * blockDim.x + threadIdx.x;
  if (i < B * H) {
    h_bf[i] = (__bf16)h0[i];
    c_st[i] = c0[i];
  }
  if (i < G) bias[i] = b_ih[i] + b_hh[i];
  if (i == 0) { bar[0] = 0u; bar[1] = 0u; }
}

// --- device-scope generation barrier (16 co-resident blocks) -----------------
__device__ __forceinline__ void grid_sync(uint32_t* cnt, uint32_t* gen) {
  __syncthreads();
  if (threadIdx.x == 0) {
    uint32_t g = __hip_atomic_load(gen, __ATOMIC_ACQUIRE, __HIP_MEMORY_SCOPE_AGENT);
    uint32_t arrived =
        __hip_atomic_fetch_add(cnt, 1u, __ATOMIC_ACQ_REL, __HIP_MEMORY_SCOPE_AGENT);
    if (arrived == NB - 1) {
      __hip_atomic_store(cnt, 0u, __ATOMIC_RELAXED, __HIP_MEMORY_SCOPE_AGENT);
      __hip_atomic_fetch_add(gen, 1u, __ATOMIC_RELEASE, __HIP_MEMORY_SCOPE_AGENT);
    } else {
      while (__hip_atomic_load(gen, __ATOMIC_ACQUIRE, __HIP_MEMORY_SCOPE_AGENT) == g)
        __builtin_amdgcn_s_sleep(2);
    }
  }
  __syncthreads();
}

// --- persistent LSTM: all 511 steps in one kernel ----------------------------
// 16 blocks x 256 threads (8 waves). Global wave gw = bid*8+w:
//   m_tile = gw>>6  (batch rows m0..m0+15), j-tile = gw&63 (16 gate columns).
// Wave computes the same 16 columns in all 4 gate quadrants:
//   n0_q = j0 + q*1024, q in {i,f,g,o}  -> 4 independent WMMA chains.
__global__ void lstm_persistent(const __bf16* __restrict__ embbf, // [511][32][1024]
                                const __bf16* __restrict__ Wcat,  // [4096][2048]
                                const float*  __restrict__ bias,  // [4096]
                                __bf16* __restrict__ h_bf,        // [32][1024]
                                float*  __restrict__ c_st,        // [32][1024]
                                float*  __restrict__ res,         // [32][511][1024]
                                float*  __restrict__ hT,
                                float*  __restrict__ cT,
                                uint32_t* __restrict__ bar) {
  __shared__ __bf16 ldsA[16 * K2];                                // 64 KB

  const int tid  = threadIdx.x;
  const int bb   = blockIdx.x;              // 0..15
  const int m0   = (bb >> 3) * 16;          // blocks 0-7: m0=0, 8-15: m0=16
  const int wave = tid >> 5;
  const int lane = tid & 31;
  const int gw   = bb * 8 + wave;           // 0..127
  const int j0   = (gw & 63) * 16;          // 64 column tiles per m

  // ISA 7.12.2 wave32 bf16 lane maps (loop-invariant)
  const int kba = (lane < 16) ? 0 : 8;      // A half-wave K-base
  const int kbb = (lane < 16) ? 0 : 16;     // unused; B loads use kba too
  const int jc  = j0 + (lane & 15);         // this lane's gate column (q=0)
  const __bf16* arow = ldsA + (lane & 15) * K2;
  const __bf16* br0  = Wcat + (size_t)(jc)         * K2;
  const __bf16* br1  = Wcat + (size_t)(jc + H)     * K2;
  const __bf16* br2  = Wcat + (size_t)(jc + 2 * H) * K2;
  const __bf16* br3  = Wcat + (size_t)(jc + 3 * H) * K2;
  const float bi = bias[jc];
  const float bf_ = bias[H + jc];
  const float bg = bias[2 * H + jc];
  const float bo = bias[3 * H + jc];
  const int rbase = m0 + ((lane < 16) ? 0 : 8);

  uint32_t* ldsw = (uint32_t*)ldsA;
  const uint32_t* hw = (const uint32_t*)h_bf;

  for (int t = 0; t < TP; ++t) {
    // ---- stage A = [h | x_t] rows m0..m0+15 into LDS (16 x 2048 bf16) ----
    const uint32_t* ew = (const uint32_t*)(embbf + (size_t)t * B * H);
#if USE_ASYNC_LDS
    // CDNA5 async copy engine: memory -> LDS directly, no VGPR round-trip.
#pragma unroll
    for (int i = 0; i < 16; ++i) {                  // 4096 b128 chunks / block
      int q  = tid + (i << 8);
      int w  = q << 2;                              // first word of chunk
      int r  = w >> 10;
      int kw = w & 1023;
      const uint32_t* src = (kw < 512) ? (hw + (m0 + r) * 512 + kw)
                                       : (ew + (m0 + r) * 512 + (kw - 512));
      __builtin_amdgcn_global_load_async_to_lds_b128(
          (v4i*)src, (v4i*)(ldsw + w), /*offset=*/0, /*cpol=*/0);
    }
    __builtin_amdgcn_s_wait_asynccnt(0);
#else
#pragma unroll
    for (int i = 0; i < 64; ++i) {                  // 16384 words
      int w  = tid + (i << 8);
      int r  = w >> 10;
      int kw = w & 1023;
      ldsw[w] = (kw < 512) ? hw[(m0 + r) * 512 + kw]
                           : ew[(m0 + r) * 512 + (kw - 512)];
    }
#endif
    __syncthreads();

    // ---- 4 independent K=2048 WMMA chains (i, f, g, o quadrants) ----
    v8f ai = {0.f,0.f,0.f,0.f,0.f,0.f,0.f,0.f};
    v8f af = ai, ag = ai, ao = ai;
    for (int kc = 0; kc < K2 / 32; ++kc) {          // 64 k-chunks
      const int k0 = kc << 5;
      __builtin_prefetch(br0 + k0 + 512, 0, 1);     // global_prefetch_b8
      __builtin_prefetch(br2 + k0 + 512, 0, 1);
      AFrag a, b0, b1, b2, b3;
#pragma unroll
      for (int v = 0; v < 8; ++v)
        a.p[v] = *(const v2bf*)(arow + k0 + ((v & 4) << 2) + kba + ((v & 3) << 1));
#pragma unroll
      for (int v = 0; v < 8; ++v) {
        b0.p[v] = *(const v2bf*)(br0 + k0 + ((v & 4) << 2) + kba + ((v & 3) << 1));
        b1.p[v] = *(const v2bf*)(br1 + k0 + ((v & 4) << 2) + kba + ((v & 3) << 1));
        b2.p[v] = *(const v2bf*)(br2 + k0 + ((v & 4) << 2) + kba + ((v & 3) << 1));
        b3.p[v] = *(const v2bf*)(br3 + k0 + ((v & 4) << 2) + kba + ((v & 3) << 1));
      }
      ai = __builtin_amdgcn_wmma_f32_16x16x32_bf16(false, a.v, false, b0.v,
                                                   (short)0, ai, false, false);
      af = __builtin_amdgcn_wmma_f32_16x16x32_bf16(false, a.v, false, b1.v,
                                                   (short)0, af, false, false);
      ag = __builtin_amdgcn_wmma_f32_16x16x32_bf16(false, a.v, false, b2.v,
                                                   (short)0, ag, false, false);
      ao = __builtin_amdgcn_wmma_f32_16x16x32_bf16(false, a.v, false, b3.v,
                                                   (short)0, ao, false, false);
    }

    // ---- fused activations + state update (register-resident) ----
    // acc element v: batch row b = rbase + v, gate column jc.
#pragma unroll
    for (int v = 0; v < 8; ++v) {
      const int brow = rbase + v;
      const int ci   = brow * H + jc;
      float si = fsigmoid(ai[v] + bi);
      float sf = fsigmoid(af[v] + bf_);
      float tg = ftanh  (ag[v] + bg);
      float so = fsigmoid(ao[v] + bo);
      float cn = sf * c_st[ci] + si * tg;
      float hn = so * ftanh(cn);
      c_st[ci] = cn;
      h_bf[ci] = (__bf16)hn;
      res[((size_t)brow * TP + t) * H + jc] = hn;
      if (t == TP - 1) { hT[ci] = hn; cT[ci] = cn; }
    }

    // ---- warm next step's x_t slice into WGP$/L2 while we sit at barrier ----
    if (t + 1 < TP) {
      const char* nx = (const char*)(embbf + (size_t)(t + 1) * B * H)
                       + (size_t)m0 * (H * 2 / 16) * 16;   // rows m0..m0+15
      __builtin_prefetch(nx + (size_t)tid * 128, 0, 1);    // 256 x 128B = 32 KB
    }

    // ---- publish h/c, then all 16 blocks re-read h next step ----
    grid_sync(bar, bar + 1);
  }
}

// ---------------------------------------------------------------------------
extern "C" void kernel_launch(void* const* d_in, const int* in_sizes, int n_in,
                              void* d_out, int out_size, void* d_ws, size_t ws_size,
                              hipStream_t stream) {
  (void)in_sizes; (void)n_in; (void)out_size; (void)ws_size;

  const int*   tgt       = (const int*)  d_in[0];
  const float* h0        = (const float*)d_in[1];
  const float* c0        = (const float*)d_in[2];
  // d_in[3] encoder_outputs, d_in[4] src_lengths: unused by the reference.
  const float* emb_table = (const float*)d_in[5];
  const float* W_ih      = (const float*)d_in[6];
  const float* W_hh      = (const float*)d_in[7];
  const float* b_ih      = (const float*)d_in[8];
  const float* b_hh      = (const float*)d_in[9];

  char* ws = (char*)d_ws;
  __bf16*   wcat  = (__bf16*)  (ws + OFF_WCAT);
  __bf16*   embbf = (__bf16*)  (ws + OFF_EMB);
  float*    bias  = (float*)   (ws + OFF_BIAS);
  __bf16*   h_bf  = (__bf16*)  (ws + OFF_HBF);
  float*    c_st  = (float*)   (ws + OFF_C);
  uint32_t* bar   = (uint32_t*)(ws + OFF_BAR);

  float* res = (float*)d_out;                       // [32][511][1024]
  float* hT  = res + (size_t)B * TP * H;            // [1][32][1024]
  float* cT  = hT + (size_t)B * H;                  // [1][32][1024]

  {
    int pairs = G * (K2 / 2);
    prep_wcat<<<(pairs + 255) / 256, 256, 0, stream>>>(W_ih, W_hh, (uint32_t*)wcat);
  }
  {
    int pairs = TP * B * (H / 2);
    prep_emb<<<(pairs + 255) / 256, 256, 0, stream>>>(tgt, emb_table, (uint32_t*)embbf);
  }
  prep_state<<<(B * H + 255) / 256, 256, 0, stream>>>(h0, c0, b_ih, b_hh,
                                                      bias, h_bf, c_st, bar);

  // all 511 recurrent steps inside one persistent kernel
  lstm_persistent<<<NB, 256, 0, stream>>>(embbf, wcat, bias, h_bf, c_st,
                                          res, hT, cT, bar);
}
